// GlobalPointer_65919158059263
// MI455X (gfx1250) — compile-verified, matrix operand order checked by reading, not run on hardware
//
#include <hip/hip_runtime.h>
#include <hip/hip_bf16.h>

#define HEADS 9
#define HEAD_SIZE 64
#define HIDDEN 768
#define BDIM 8
#define SDIM 1024
#define NOUT (HEADS * HEAD_SIZE * 2) /* 1152 */
#define MROWS (BDIM * SDIM)          /* 8192 */
#define NEGV 1000000000000.0f

typedef __attribute__((ext_vector_type(16))) __bf16 v16bf;
typedef __attribute__((ext_vector_type(8)))  float  v8f;
typedef __attribute__((ext_vector_type(4)))  float  v4f;
typedef __attribute__((ext_vector_type(4)))  unsigned int v4u;

union Frag16 {
    v16bf  v;
    __bf16 e[16];
    v4u    q[2]; // 32 bytes = 16 bf16
};

// Native converts: clang should lower to v_cvt_*bf16_f32 (RNE), far cheaper
// than the integer-rounding trick (which cost ~4 VALU/element).
static __device__ __forceinline__ __bf16 f2bf(float f) { return (__bf16)f; }

// lane <-> lane^1 swap via DPP quad_perm([1,0,3,2]); co-issuable VALU,
// avoids ds_bpermute + DScnt waits.
static __device__ __forceinline__ float lane_xor1(float v) {
    int i = __builtin_bit_cast(int, v);
    i = __builtin_amdgcn_mov_dpp(i, 0xB1, 0xF, 0xF, true);
    return __builtin_bit_cast(float, i);
}

// ---- W: [HIDDEN][NOUT] f32 -> Wt: [NOUT][HIDDEN] bf16 ---------------------
__global__ void cvt_transpose_W(const float* __restrict__ W,
                                __bf16* __restrict__ Wt) {
    int i = blockIdx.x * blockDim.x + threadIdx.x;
    if (i < HIDDEN * NOUT) {
        int k = i / NOUT, n = i % NOUT;
        Wt[(size_t)n * HIDDEN + k] = f2bf(W[i]);
    }
}

// ---- Fused GEMM1 + bias + RoPE -------------------------------------------
// inputs: [MROWS][HIDDEN] f32 (converted to bf16 in-register)
// Wt:     [NOUT][HIDDEN] bf16
// Writes rotated Q/K directly as bf16 [B,H,S,64].
__global__ __launch_bounds__(128)
void gemm1_rope_kernel(const float* __restrict__ inputs,
                       const __bf16* __restrict__ Wt,
                       const float* __restrict__ bias,
                       __bf16* __restrict__ Qb, __bf16* __restrict__ Kb) {
    const int lane = threadIdx.x & 31;
    const int wave = threadIdx.x >> 5;      // 0..3 -> 2x2 wave grid
    const int row0 = blockIdx.x * 128 + (wave >> 1) * 64;
    const int col0 = blockIdx.y * 128 + (wave & 1) * 64;
    const int m16  = lane & 15;
    const int half = lane >> 4;

    // accumulators initialized with bias (same bias across the 8 rows/VGPR)
    v8f c[4][4];
    for (int i = 0; i < 4; ++i)
        for (int j = 0; j < 4; ++j) {
            float bv = bias[col0 + j * 16 + m16];
            v8f t;
            for (int r = 0; r < 8; ++r) t[r] = bv;
            c[i][j] = t;
        }

    for (int k0 = 0; k0 < HIDDEN; k0 += 32) {
        Frag16 a[4], bm[4];
        for (int i = 0; i < 4; ++i) {
            const float* p = inputs + (size_t)(row0 + i * 16 + m16) * HIDDEN + k0 + half * 8;
            v4f f0 = __builtin_nontemporal_load((const v4f*)p);
            v4f f1 = __builtin_nontemporal_load((const v4f*)(p + 4));
            v4f f2 = __builtin_nontemporal_load((const v4f*)(p + 16));
            v4f f3 = __builtin_nontemporal_load((const v4f*)(p + 20));
            for (int t = 0; t < 4; ++t) {
                a[i].e[t]      = f2bf(f0[t]);
                a[i].e[4 + t]  = f2bf(f1[t]);
                a[i].e[8 + t]  = f2bf(f2[t]);
                a[i].e[12 + t] = f2bf(f3[t]);
            }
        }
        for (int j = 0; j < 4; ++j) {
            const __bf16* p = Wt + (size_t)(col0 + j * 16 + m16) * HIDDEN + k0 + half * 8;
            bm[j].q[0] = *(const v4u*)p;
            bm[j].q[1] = *(const v4u*)(p + 16);
        }
        for (int i = 0; i < 4; ++i)
            for (int j = 0; j < 4; ++j)
                c[i][j] = __builtin_amdgcn_wmma_f32_16x16x32_bf16(
                    false, a[i].v, false, bm[j].v, (short)0, c[i][j], false, false);
    }

    // Fused RoPE epilogue.
    // Column dd = col % 128 within head h = col/128; q half dd<64, k half dd>=64.
    // Pair (2i,2i+1) lives in adjacent lanes -> partner via DPP quad-perm swap.
    // cos/sin index = dd % 32; angle = s * 10000^(-(dd%32)/32).
    const float lg = -9.2103403719761836f / 32.0f; // -ln(10000)/32
    for (int j = 0; j < 4; ++j) {
        int col  = col0 + j * 16 + m16;
        int h    = col >> 7;
        int dd   = col & 127;
        float fr = __expf((float)(dd & 31) * lg);
        bool odd = (dd & 1) != 0;
        int  d   = (dd < 64) ? dd : (dd - 64);
        __bf16* dst = (dd < 64) ? Qb : Kb;
        for (int i = 0; i < 4; ++i) {
            for (int r = 0; r < 8; ++r) {
                int grow = row0 + i * 16 + half * 8 + r; // = b*S + s
                int s    = grow & (SDIM - 1);
                int b    = grow >> 10;
                float ang = (float)s * fr;
                float ca = __cosf(ang), sa = __sinf(ang);
                float val = c[i][j][r];
                float par = lane_xor1(val); // value at column dd^1, same row
                float rot = odd ? (val * ca + par * sa) : (val * ca - par * sa);
                size_t base = (((size_t)b * HEADS + h) * SDIM + s) * HEAD_SIZE;
                dst[base + d] = f2bf(rot);
            }
        }
    }
}

// ---- GEMM2: logits[b,h] = q @ k^T, fused mask + scale, NT stores ----------
__global__ __launch_bounds__(128)
void attn_logits_kernel(const __bf16* __restrict__ Q, const __bf16* __restrict__ Km,
                        const float* __restrict__ mask, float* __restrict__ out) {
    const int bh = blockIdx.z;
    const int b  = bh / HEADS;
    const __bf16* q = Q  + (size_t)bh * SDIM * HEAD_SIZE;
    const __bf16* k = Km + (size_t)bh * SDIM * HEAD_SIZE;

    const int lane = threadIdx.x & 31;
    const int wave = threadIdx.x >> 5;
    const int row0 = blockIdx.x * 128 + (wave >> 1) * 64;
    const int col0 = blockIdx.y * 128 + (wave & 1) * 64;
    const int m16  = lane & 15;
    const int half = lane >> 4;

    v8f c[4][4] = {};
    for (int k0 = 0; k0 < HEAD_SIZE; k0 += 32) {
        Frag16 a[4], bm[4];
        for (int i = 0; i < 4; ++i) {
            const __bf16* p = q + (size_t)(row0 + i * 16 + m16) * HEAD_SIZE + k0 + half * 8;
            a[i].q[0] = *(const v4u*)p;
            a[i].q[1] = *(const v4u*)(p + 16);
        }
        for (int j = 0; j < 4; ++j) {
            const __bf16* p = k + (size_t)(col0 + j * 16 + m16) * HEAD_SIZE + k0 + half * 8;
            bm[j].q[0] = *(const v4u*)p;
            bm[j].q[1] = *(const v4u*)(p + 16);
        }
        for (int i = 0; i < 4; ++i)
            for (int j = 0; j < 4; ++j)
                c[i][j] = __builtin_amdgcn_wmma_f32_16x16x32_bf16(
                    false, a[i].v, false, bm[j].v, (short)0, c[i][j], false, false);
    }

    // Epilogue: out = (c*pad - NEG*(1-pad) - (m>n)*NEG) * scale
    //         = fma(c, pad*scale, beta) with beta folded per element.
    const float scale  = 0.125f;         // 1/sqrt(64)
    const float negsc  = -NEGV * scale;  // -NEG/8
    float* outp = out + (size_t)bh * SDIM * SDIM;
    const float* mrow = mask + (size_t)b * SDIM;
    const int n = col0 + m16; // per-lane base column (add j*16 inside)
    for (int i = 0; i < 4; ++i) {
        const int mbase = row0 + i * 16 + half * 8;
        float pm[8];
        for (int r = 0; r < 8; ++r) pm[r] = mrow[mbase + r];
        for (int j = 0; j < 4; ++j) {
            int nj = n + j * 16;
            float mn = mrow[nj];
            for (int r = 0; r < 8; ++r) {
                int m = mbase + r;
                float pad   = pm[r] * mn;
                float alpha = pad * scale;
                float beta  = negsc * (1.0f - pad);
                if (m > nj) beta += negsc; // tril(ones,-1) causal term
                // streaming 302MB result: non-temporal so Q/K stay L2-resident
                __builtin_nontemporal_store(fmaf(c[i][j][r], alpha, beta),
                                            &outp[(size_t)m * SDIM + nj]);
            }
        }
    }
}

extern "C" void kernel_launch(void* const* d_in, const int* in_sizes, int n_in,
                              void* d_out, int out_size, void* d_ws, size_t ws_size,
                              hipStream_t stream) {
    const float* inputs = (const float*)d_in[0]; // [B,S,HIDDEN]
    const float* mask   = (const float*)d_in[1]; // [B,S]
    const float* W      = (const float*)d_in[2]; // [HIDDEN, NOUT]
    const float* bias   = (const float*)d_in[3]; // [NOUT]
    float* out = (float*)d_out;                  // [B,HEADS,S,S]

    // workspace layout (bf16 elements): Wt | Q | K  (~20.6 MB total)
    __bf16* Wt_bf = (__bf16*)d_ws;                                  // NOUT*HIDDEN
    __bf16* Qb    = Wt_bf + (size_t)NOUT * HIDDEN;                  // B*H*S*64
    __bf16* Kb    = Qb    + (size_t)BDIM * HEADS * SDIM * HEAD_SIZE;

    int nW = HIDDEN * NOUT;
    cvt_transpose_W<<<(nW + 255) / 256, 256, 0, stream>>>(W, Wt_bf);

    dim3 g1(MROWS / 128, NOUT / 128); // 64 x 9
    gemm1_rope_kernel<<<g1, 128, 0, stream>>>(inputs, Wt_bf, bias, Qb, Kb);

    dim3 g2(SDIM / 128, SDIM / 128, BDIM * HEADS); // 8 x 8 x 72
    attn_logits_kernel<<<g2, 128, 0, stream>>>(Qb, Kb, mask, out);
}
